// DCNConvModule_19928648253490
// MI455X (gfx1250) — matrix operand order repaired and checked by reading, not compiled, not run on hardware
//
#include <hip/hip_runtime.h>

typedef __attribute__((ext_vector_type(16))) __bf16 v16bf;
typedef __attribute__((ext_vector_type(8)))  float  v8f;

// Problem constants
// B=4, C=256, H=W=64, Cout=256, K=3 (K2=9), KK = C*K2 = 2304, HW = 4096

static __device__ __forceinline__ unsigned short f2bf(float f) {
  unsigned int u = __float_as_uint(f);
  unsigned int r = u + 0x7FFFu + ((u >> 16) & 1u);   // round-to-nearest-even
  return (unsigned short)(r >> 16);
}

// ---------------------------------------------------------------------------
// Kernel 1: offset conv  om[b, 27, h, w] = conv3x3(x, w_off) + b_off
// grid (16, 4) x 256 threads; thread = (h = bx*4 + tid/64, w = tid%64)
// ---------------------------------------------------------------------------
__global__ void k_offset_conv(const float* __restrict__ x,
                              const float* __restrict__ w_off,
                              const float* __restrict__ b_off,
                              float* __restrict__ om) {
  const int tid = threadIdx.x;
  const int w = tid & 63;
  const int h = blockIdx.x * 4 + (tid >> 6);
  const int b = blockIdx.y;

  float acc[27];
#pragma unroll
  for (int i = 0; i < 27; ++i) acc[i] = 0.f;

  const float* xb = x + (size_t)b * 256 * 4096;
  for (int c = 0; c < 256; ++c) {
    float xv[9];
#pragma unroll
    for (int ki = 0; ki < 3; ++ki) {
      const int hh = h - 1 + ki;
      const bool hv = (hh >= 0) && (hh < 64);
#pragma unroll
      for (int kj = 0; kj < 3; ++kj) {
        const int ww = w - 1 + kj;
        const bool wv = (ww >= 0) && (ww < 64);
        xv[ki * 3 + kj] = (hv && wv) ? xb[c * 4096 + hh * 64 + ww] : 0.f;
      }
    }
    const float* wb = w_off + c * 9;   // + oc*2304 + k
#pragma unroll
    for (int k = 0; k < 9; ++k) {
      const float xk = xv[k];
#pragma unroll
      for (int oc = 0; oc < 27; ++oc)
        acc[oc] = fmaf(xk, wb[oc * 2304 + k], acc[oc]);
    }
  }
#pragma unroll
  for (int oc = 0; oc < 27; ++oc)
    om[((size_t)(b * 27 + oc)) * 4096 + h * 64 + w] = acc[oc] + b_off[oc];
}

// ---------------------------------------------------------------------------
// Kernel 2: bilinear sample * mask -> bf16 A matrix
// A[(b*4096 + h*64 + w)][kk*256 + c]   (K-dim layout: kk-major, c-minor)
// grid (H=64, K2=9, B=4) x 64 threads (thread = w)
// ---------------------------------------------------------------------------
__global__ void k_sample_pack(const float* __restrict__ x,
                              const float* __restrict__ om,
                              unsigned short* __restrict__ A) {
  const int w  = threadIdx.x;
  const int h  = blockIdx.x;
  const int kk = blockIdx.y;
  const int b  = blockIdx.z;
  const int ki = kk / 3, kj = kk % 3;

  const size_t omb = (size_t)b * 27 * 4096 + (size_t)h * 64 + w;
  const float oy = om[omb + (size_t)(2 * kk) * 4096];
  const float ox = om[omb + (size_t)(2 * kk + 1) * 4096];
  const float mv = om[omb + (size_t)(18 + kk) * 4096];
  const float mask = 1.f / (1.f + __expf(-mv));

  const float ys = (float)(h - 1 + ki) + oy;
  const float xs = (float)(w - 1 + kj) + ox;
  const float y0 = floorf(ys), x0 = floorf(xs);
  const float wy = ys - y0, wx = xs - x0;

  int   idx[4];
  float wt[4];
#pragma unroll
  for (int j = 0; j < 4; ++j) {
    const float yc = y0 + (float)(j >> 1);
    const float xc = x0 + (float)(j & 1);
    const bool valid = (yc >= 0.f) && (yc <= 63.f) && (xc >= 0.f) && (xc <= 63.f);
    const int yi = (int)fminf(fmaxf(yc, 0.f), 63.f);
    const int xi = (int)fminf(fmaxf(xc, 0.f), 63.f);
    idx[j] = yi * 64 + xi;
    const float wyj = (j >> 1) ? wy : (1.f - wy);
    const float wxj = (j & 1) ? wx : (1.f - wx);
    wt[j] = valid ? (wyj * wxj * mask) : 0.f;
  }

  const float* xb = x + (size_t)b * 256 * 4096;
  unsigned short* Ab = A + ((size_t)(b * 4096 + h * 64 + w)) * 2304 + kk * 256;
#pragma unroll 4
  for (int c = 0; c < 256; ++c) {
    const float* xc_ = xb + (size_t)c * 4096;
    const float v = wt[0] * xc_[idx[0]] + wt[1] * xc_[idx[1]] +
                    wt[2] * xc_[idx[2]] + wt[3] * xc_[idx[3]];
    Ab[c] = f2bf(v);
  }
}

// ---------------------------------------------------------------------------
// Kernel 3: pack weight to bf16, K-major layout matching A:
// Wb[o][kk*256 + c] = bf16(weight[o][c][kk]),  2304 blocks x 256 threads
// ---------------------------------------------------------------------------
__global__ void k_pack_w(const float* __restrict__ weight,
                         unsigned short* __restrict__ Wb) {
  const int i  = blockIdx.x * 256 + threadIdx.x;     // < 256*2304
  const int o  = i / 2304;
  const int r  = i - o * 2304;
  const int kk = r >> 8;
  const int c  = r & 255;
  Wb[(size_t)o * 2304 + r] = f2bf(weight[(size_t)o * 2304 + c * 9 + kk]);
}

// ---------------------------------------------------------------------------
// Kernel 4: bf16 WMMA GEMM.  D[m][n] = sum_k A[b][m][k] * Wb[n][k] + bias[n]
// stored transposed into conv[b][n][m] (NCHW).
// grid (32 Mblk, 2 Nblk, 4 b) x 256 threads; 8 waves = 4(M) x 2(N);
// wave tile 32M x 64N (2x4 accumulators); block tile 128 x 128; K step 32.
// ---------------------------------------------------------------------------
__global__ void k_gemm(const unsigned short* __restrict__ A,
                       const unsigned short* __restrict__ Wb,
                       const float* __restrict__ bias,
                       float* __restrict__ conv) {
  const int lane = threadIdx.x & 31;
  const int wave = threadIdx.x >> 5;
  const int wm = wave & 3, wn = wave >> 2;
  const int b  = blockIdx.z;
  const int m0 = blockIdx.x * 128 + wm * 32;
  const int n0 = blockIdx.y * 128 + wn * 64;
  const int lr   = lane & 15;
  const int half = lane >> 4;

  v8f acc[2][4];
  const v8f vz = {0.f, 0.f, 0.f, 0.f, 0.f, 0.f, 0.f, 0.f};
#pragma unroll
  for (int mi = 0; mi < 2; ++mi)
#pragma unroll
    for (int ni = 0; ni < 4; ++ni) acc[mi][ni] = vz;

  const unsigned short* Abase = A + (size_t)(b * 4096) * 2304;
  const unsigned short* arow[2];
#pragma unroll
  for (int mi = 0; mi < 2; ++mi)
    arow[mi] = Abase + (size_t)(m0 + mi * 16 + lr) * 2304;
  const unsigned short* brow[4];
#pragma unroll
  for (int ni = 0; ni < 4; ++ni)
    brow[ni] = Wb + (size_t)(n0 + ni * 16 + lr) * 2304;

  union Frag { uint4 q[2]; v16bf v; };

  for (int k0 = 0; k0 < 2304; k0 += 32) {
    Frag a[2], bf[4];
    // A 16x32 bf16: lane<16 -> row M=lr, K {0..7,16..23}; lane>=16 -> K {8..15,24..31}
#pragma unroll
    for (int mi = 0; mi < 2; ++mi) {
      const unsigned short* p = arow[mi] + k0 + half * 8;
      a[mi].q[0] = *(const uint4*)(p);
      a[mi].q[1] = *(const uint4*)(p + 16);
    }
    // B 32x16 bf16: lane<16 -> col N=lr, K {0..7,16..23}; lane>=16 -> K {8..15,24..31}
#pragma unroll
    for (int ni = 0; ni < 4; ++ni) {
      const unsigned short* p = brow[ni] + k0 + half * 8;
      bf[ni].q[0] = *(const uint4*)(p);
      bf[ni].q[1] = *(const uint4*)(p + 16);
    }
#pragma unroll
    for (int mi = 0; mi < 2; ++mi)
#pragma unroll
      for (int ni = 0; ni < 4; ++ni)
        acc[mi][ni] = __builtin_amdgcn_wmma_f32_16x16x32_bf16(
            false, a[mi].v, false, bf[ni].v, (short)0, acc[mi][ni], false, false);
  }

  float bv[4];
#pragma unroll
  for (int ni = 0; ni < 4; ++ni) bv[ni] = bias[n0 + ni * 16 + lr];

  float* cb = conv + (size_t)b * 256 * 4096;
#pragma unroll
  for (int mi = 0; mi < 2; ++mi) {
    const int mbase = m0 + mi * 16 + half * 8;   // C/D: lanes>=16 hold M=8..15
#pragma unroll
    for (int ni = 0; ni < 4; ++ni) {
      const int n = n0 + ni * 16 + lr;
      float* cp = cb + (size_t)n * 4096 + mbase;
#pragma unroll
      for (int r = 0; r < 8; ++r)
        cp[r] = acc[mi][ni][r] + bv[ni];
    }
  }
}

// ---------------------------------------------------------------------------
// Kernel 5/6: deterministic two-stage per-batch sum / sumsq reduction
// ---------------------------------------------------------------------------
__global__ void k_partial(const float* __restrict__ conv, float* __restrict__ part) {
  __shared__ float ss[256], sq[256];
  const int b = blockIdx.y, t = threadIdx.x;
  const float* p = conv + ((size_t)b << 20) + (size_t)blockIdx.x * 4096;
  float s = 0.f, q = 0.f;
#pragma unroll
  for (int j = 0; j < 16; ++j) {
    const float v = p[t + j * 256];
    s += v; q += v * v;
  }
  ss[t] = s; sq[t] = q;
  __syncthreads();
  for (int off = 128; off > 0; off >>= 1) {
    if (t < off) { ss[t] += ss[t + off]; sq[t] += sq[t + off]; }
    __syncthreads();
  }
  if (t == 0) {
    part[(b * 256 + blockIdx.x) * 2]     = ss[0];
    part[(b * 256 + blockIdx.x) * 2 + 1] = sq[0];
  }
}

__global__ void k_final(const float* __restrict__ part, float* __restrict__ stats) {
  __shared__ float ss[256], sq[256];
  const int b = blockIdx.x, t = threadIdx.x;
  ss[t] = part[(b * 256 + t) * 2];
  sq[t] = part[(b * 256 + t) * 2 + 1];
  __syncthreads();
  for (int off = 128; off > 0; off >>= 1) {
    if (t < off) { ss[t] += ss[t + off]; sq[t] += sq[t + off]; }
    __syncthreads();
  }
  if (t == 0) { stats[b * 2] = ss[0]; stats[b * 2 + 1] = sq[0]; }
}

// ---------------------------------------------------------------------------
// Kernel 7: normalize + affine + relu  (float4 vectorized)
// ---------------------------------------------------------------------------
__global__ void k_norm(const float* __restrict__ conv, const float* __restrict__ stats,
                       const float* __restrict__ gamma, const float* __restrict__ beta,
                       float* __restrict__ out) {
  const int i4 = blockIdx.x * 256 + threadIdx.x;
  const size_t base = (size_t)i4 * 4;
  const int b = (int)(base >> 20);
  const int o = (int)((base >> 12) & 255);
  const float inN = 1.f / 1048576.f;
  const float mu  = stats[b * 2] * inN;
  const float var = stats[b * 2 + 1] * inN - mu * mu;
  const float inv = rsqrtf(var + 1e-5f);
  const float g = gamma[o] * inv, bt = beta[o];
  const float4 v = *(const float4*)(conv + base);
  float4 r;
  r.x = fmaxf((v.x - mu) * g + bt, 0.f);
  r.y = fmaxf((v.y - mu) * g + bt, 0.f);
  r.z = fmaxf((v.z - mu) * g + bt, 0.f);
  r.w = fmaxf((v.w - mu) * g + bt, 0.f);
  *(float4*)(out + base) = r;
}

// ---------------------------------------------------------------------------
extern "C" void kernel_launch(void* const* d_in, const int* in_sizes, int n_in,
                              void* d_out, int out_size, void* d_ws, size_t ws_size,
                              hipStream_t stream) {
  const float* x      = (const float*)d_in[0];
  const float* w_off  = (const float*)d_in[1];
  const float* b_off  = (const float*)d_in[2];
  const float* weight = (const float*)d_in[3];
  const float* bias   = (const float*)d_in[4];
  const float* gamma  = (const float*)d_in[5];
  const float* beta   = (const float*)d_in[6];
  float* out = (float*)d_out;

  char* ws = (char*)d_ws;
  float*          om    = (float*)(ws);                          // 1,769,472 B
  unsigned short* A     = (unsigned short*)(ws + 1769472);       // 75,497,472 B
  unsigned short* Wb    = (unsigned short*)(ws + 77266944);      // 1,179,648 B
  float*          conv  = (float*)(ws + 78446592);               // 16,777,216 B
  float*          part  = (float*)(ws + 95223808);               // 8,192 B
  float*          stats = (float*)(ws + 95232000);               // 32 B

  k_offset_conv<<<dim3(16, 4), 256, 0, stream>>>(x, w_off, b_off, om);
  k_sample_pack<<<dim3(64, 9, 4), 64, 0, stream>>>(x, om, A);
  k_pack_w<<<2304, 256, 0, stream>>>(weight, Wb);
  k_gemm<<<dim3(32, 2, 4), 256, 0, stream>>>(A, Wb, bias, conv);
  k_partial<<<dim3(256, 4), 256, 0, stream>>>(conv, part);
  k_final<<<4, 256, 0, stream>>>(part, stats);
  k_norm<<<4096, 256, 0, stream>>>(conv, stats, gamma, beta, out);
}
